// GlobalSceneEncoder_23338852286546
// MI455X (gfx1250) — matrix operand order repaired and checked by reading, not compiled, not run on hardware
//
#include <hip/hip_runtime.h>
#include <hip/hip_bf16.h>

#define CDIM 384
#define KNN  32
#define BB   4

typedef __attribute__((ext_vector_type(16))) _Float16 v16h;
typedef __attribute__((ext_vector_type(8)))  float    v8f;

// ---------------------------------------------------------------------------
// Block-wide sum over CDIM(=384) threads. s must be shared float[CDIM].
// ---------------------------------------------------------------------------
__device__ __forceinline__ float blockSum384(float v, float* s) {
  int t = threadIdx.x;
  __syncthreads();
  s[t] = v;
  __syncthreads();
  for (int st = 192; st >= 3; st >>= 1) {   // 192,96,48,24,12,6,3
    if (t < st) s[t] += s[t + st];
    __syncthreads();
  }
  return s[0] + s[1] + s[2];
}

// ---------------------------------------------------------------------------
// Split pts (B,N,3+C) into pos (B,N,3) and x (B,N,C). grid = B*N, block = CDIM
// ---------------------------------------------------------------------------
__global__ void split_pts_kernel(const float* __restrict__ pts,
                                 float* __restrict__ pos,
                                 float* __restrict__ x) {
  int row = blockIdx.x;
  int c   = threadIdx.x;
  x[(size_t)row * CDIM + c] = pts[(size_t)row * (CDIM + 3) + 3 + c];
  if (c < 3) pos[(size_t)row * 3 + c] = pts[(size_t)row * (CDIM + 3) + c];
}

// ---------------------------------------------------------------------------
// WMMA GEMM: Y[R x 384] = act(X[R x 384] @ W[384 x 384] + bias)
// block = 128 threads (4 wave32s) computes a 32x64 output tile:
//   each wave owns one 16-col slab and TWO 16x16 accumulators (two M-tiles),
//   issuing 2 back-to-back v_wmma per K-step; K-loop fully unrolled (12 it).
// grid = (ceil(R/32), 384/64)
// ---------------------------------------------------------------------------
__global__ void wmma_gemm_kernel(const float* __restrict__ X,
                                 const float* __restrict__ W,
                                 const float* __restrict__ bias,
                                 float* __restrict__ Y, int R, int act) {
  __shared__ float sA[32 * 32];
  __shared__ float sB[32 * 64];
  int tid  = threadIdx.x;
  int wave = tid >> 5;
  int lane = tid & 31;
  int m    = lane & 15;
  int hi   = lane >> 4;
  int row0 = blockIdx.x * 32;
  int wc0  = blockIdx.y * 64;
  v8f acc0 = {};
  v8f acc1 = {};
#pragma unroll
  for (int k0 = 0; k0 < CDIM; k0 += 32) {
    __syncthreads();
    for (int i = tid; i < 32 * 32; i += 128) {          // X tile, coalesced
      int rr = i >> 5, kk = i & 31;
      int gr = row0 + rr;
      sA[i] = (gr < R) ? X[(size_t)gr * CDIM + k0 + kk] : 0.f;
    }
    for (int i = tid; i < 32 * 64; i += 128) {          // W tile, coalesced
      int kk = i >> 6, cc2 = i & 63;
      sB[i] = W[(size_t)(k0 + kk) * CDIM + wc0 + cc2];
    }
    if (k0 + 32 < CDIM)                                  // global_prefetch_b8
      __builtin_prefetch(&W[(size_t)(k0 + 32) * CDIM + wc0 + (tid & 63)], 0, 1);
    __syncthreads();
    // Build fragments per CDNA5 7.12.2 layouts, fp32 -> f16
    v16h a0, a1, b;
#pragma unroll
    for (int e = 0; e < 16; ++e) {
      int ka = ((e & 8) ? (e + 8) : e) + hi * 8;         // A: 16x32, row m
      a0[e] = (_Float16)sA[m * 32 + ka];
      a1[e] = (_Float16)sA[(m + 16) * 32 + ka];
      int kb = ((e & 8) ? (e + 8) : e) + hi * 8;         // B: 32x16, col m
      b[e] = (_Float16)sB[kb * 64 + wave * 16 + m];
    }
    acc0 = __builtin_amdgcn_wmma_f32_16x16x32_f16(false, a0, false, b,
                                                  (short)0, acc0, false, false);
    acc1 = __builtin_amdgcn_wmma_f32_16x16x32_f16(false, a1, false, b,
                                                  (short)0, acc1, false, false);
  }
  int gc = wc0 + wave * 16 + m;
  float bv = bias[gc];
#pragma unroll
  for (int v = 0; v < 8; ++v) {                          // C/D: M = v + hi*8
    int gr0 = row0 + v + hi * 8;
    if (gr0 < R) {
      float val = acc0[v] + bv;
      if (act) val = fmaxf(val, 0.f);
      Y[(size_t)gr0 * CDIM + gc] = val;
    }
    int gr1 = row0 + 16 + v + hi * 8;
    if (gr1 < R) {
      float val = acc1[v] + bv;
      if (act) val = fmaxf(val, 0.f);
      Y[(size_t)gr1 * CDIM + gc] = val;
    }
  }
}

// ---------------------------------------------------------------------------
// Farthest point sampling, one workgroup (256 thr) per batch. idx: local ids.
// ---------------------------------------------------------------------------
__global__ void fps_kernel(const float* __restrict__ pos, int* __restrict__ idx,
                           int Ncur, int M) {
  __shared__ float sv[256];
  __shared__ int   si[256];
  __shared__ float spt[3];
  int b = blockIdx.x;
  int t = threadIdx.x;
  const float* P = pos + (size_t)b * Ncur * 3;
  float p0x = P[0], p0y = P[1], p0z = P[2];
  float mind[16];                                        // 4096/256 max
#pragma unroll
  for (int j = 0; j < 16; ++j) {
    int gi = j * 256 + t;
    if (gi < Ncur) {
      float dx = P[gi * 3] - p0x, dy = P[gi * 3 + 1] - p0y, dz = P[gi * 3 + 2] - p0z;
      mind[j] = dx * dx + dy * dy + dz * dz;
    } else mind[j] = -1.f;                               // never selected
  }
  if (t == 0) idx[b * M] = 0;
  for (int it = 1; it < M; ++it) {
    float bv = -2.f; int bi = 0x7fffffff;
#pragma unroll
    for (int j = 0; j < 16; ++j) {                       // strict > : lowest gi wins ties
      int gi = j * 256 + t;
      if (mind[j] > bv) { bv = mind[j]; bi = gi; }
    }
    sv[t] = bv; si[t] = bi;
    __syncthreads();
    for (int st = 128; st > 0; st >>= 1) {
      if (t < st) {
        if (sv[t + st] > sv[t] || (sv[t + st] == sv[t] && si[t + st] < si[t])) {
          sv[t] = sv[t + st]; si[t] = si[t + st];
        }
      }
      __syncthreads();
    }
    int nxt = si[0];
    if (t == 0) {
      idx[b * M + it] = nxt;
      spt[0] = P[nxt * 3]; spt[1] = P[nxt * 3 + 1]; spt[2] = P[nxt * 3 + 2];
    }
    __syncthreads();
    float nx = spt[0], ny = spt[1], nz = spt[2];
#pragma unroll
    for (int j = 0; j < 16; ++j) {
      int gi = j * 256 + t;
      if (gi < Ncur) {
        float dx = P[gi * 3] - nx, dy = P[gi * 3 + 1] - ny, dz = P[gi * 3 + 2] - nz;
        float d = dx * dx + dy * dy + dz * dz;
        mind[j] = fminf(mind[j], d);
      }
    }
    __syncthreads();
  }
}

// ---------------------------------------------------------------------------
// Gather x_d, pos_d at sampled indices. grid = B*M, block = CDIM
// ---------------------------------------------------------------------------
__global__ void gather_kernel(const float* __restrict__ x, const float* __restrict__ pos,
                              const int* __restrict__ idx, float* __restrict__ xd,
                              float* __restrict__ posd, int M, int Ncur) {
  int cf = blockIdx.x;
  int b  = cf / M;
  int c  = threadIdx.x;
  int src = b * Ncur + idx[cf];
  xd[(size_t)cf * CDIM + c] = x[(size_t)src * CDIM + c];
  if (c < 3) posd[cf * 3 + c] = pos[src * 3 + c];
}

// ---------------------------------------------------------------------------
// Stable K-smallest (matches lax.top_k tie-break). 1 thread per center.
// ---------------------------------------------------------------------------
__global__ void knn_kernel(const float* __restrict__ pos, const float* __restrict__ posd,
                           int* __restrict__ nbr, int* __restrict__ validm,
                           int NC, int Ncur, int M, float r2) {
  int cf = blockIdx.x * blockDim.x + threadIdx.x;
  if (cf >= NC) return;
  int b = cf / M;
  const float* P = pos + (size_t)b * Ncur * 3;
  float cx = posd[cf * 3], cy = posd[cf * 3 + 1], cz = posd[cf * 3 + 2];
  float bd[KNN]; int bi[KNN];
#pragma unroll
  for (int k = 0; k < KNN; ++k) { bd[k] = 3.0e38f; bi[k] = 0; }
  for (int j = 0; j < Ncur; ++j) {
    float dx = P[j * 3] - cx, dy = P[j * 3 + 1] - cy, dz = P[j * 3 + 2] - cz;
    float d2 = dx * dx + dy * dy + dz * dz;
    d2 = (d2 <= r2) ? d2 : 1.0e9f;                      // BIG mask
    if (d2 < bd[KNN - 1]) {                              // strict < keeps earlier idx
      int p = KNN - 1;
      while (p > 0 && d2 < bd[p - 1]) { bd[p] = bd[p - 1]; bi[p] = bi[p - 1]; --p; }
      bd[p] = d2; bi[p] = j;
    }
  }
  for (int k = 0; k < KNN; ++k) {
    nbr[(size_t)cf * KNN + k]    = b * Ncur + bi[k];     // flat into (B*Ncur)
    validm[(size_t)cf * KNN + k] = (bd[k] < 5.0e8f) ? 1 : 0;
  }
}

// ---------------------------------------------------------------------------
// h1 = relu((pos_d - pos_j) @ w1 + b1). grid = rows (chunk), block = CDIM
// ---------------------------------------------------------------------------
__global__ void posmlp1_kernel(const float* __restrict__ pos, const float* __restrict__ posd,
                               const int* __restrict__ nbr, const float* __restrict__ w1,
                               const float* __restrict__ b1, float* __restrict__ H,
                               int centerBase) {
  int r  = blockIdx.x;
  int cf = centerBase + (r >> 5);
  int nj = nbr[(size_t)cf * KNN + (r & 31)];
  float rx = posd[cf * 3]     - pos[nj * 3];
  float ry = posd[cf * 3 + 1] - pos[nj * 3 + 1];
  float rz = posd[cf * 3 + 2] - pos[nj * 3 + 2];
  int c = threadIdx.x;
  float h = rx * w1[c] + ry * w1[CDIM + c] + rz * w1[2 * CDIM + c] + b1[c];
  H[(size_t)r * CDIM + c] = fmaxf(h, 0.f);
}

// ---------------------------------------------------------------------------
// a_in = dterm[center] - S[nbr] + delta. grid = rows, block = CDIM
// ---------------------------------------------------------------------------
__global__ void combine_kernel(const float* __restrict__ dterm, const float* __restrict__ S,
                               const int* __restrict__ nbr, const float* __restrict__ delta,
                               float* __restrict__ A, int centerBase) {
  int r  = blockIdx.x;
  int cf = centerBase + (r >> 5);
  int nj = nbr[(size_t)cf * KNN + (r & 31)];
  int c  = threadIdx.x;
  A[(size_t)r * CDIM + c] = dterm[(size_t)cf * CDIM + c]
                          - S[(size_t)nj * CDIM + c]
                          + delta[(size_t)r * CDIM + c];
}

// ---------------------------------------------------------------------------
// Softmax over K per channel, aggregate (L[nbr]+delta), LayerNorm over C.
// grid = nc centers (chunk), block = CDIM
// ---------------------------------------------------------------------------
__global__ void attn_agg_ln_kernel(const float* __restrict__ alpha, const float* __restrict__ delta,
                                   const float* __restrict__ Lb, const int* __restrict__ nbr,
                                   const int* __restrict__ validm, const float* __restrict__ g,
                                   const float* __restrict__ beta, float* __restrict__ Xout,
                                   int centerBase) {
  __shared__ float s[CDIM];
  int cf = centerBase + blockIdx.x;
  int c  = threadIdx.x;
  int rbase = blockIdx.x * KNN;                          // row within chunk
  float av[KNN];
  float mx = -1.f;
#pragma unroll
  for (int k = 0; k < KNN; ++k) {                        // relu output >= 0 -> -1 = invalid
    float a = validm[(size_t)cf * KNN + k] ? alpha[(size_t)(rbase + k) * CDIM + c] : -1.f;
    av[k] = a;
    mx = fmaxf(mx, a);
  }
  float ssum = 0.f;
#pragma unroll
  for (int k = 0; k < KNN; ++k) {
    float e = (av[k] >= 0.f) ? __expf(av[k] - mx) : 0.f;
    av[k] = e;
    ssum += e;
  }
  float inv = 1.f / ssum;
  float out = 0.f;
  for (int k = 0; k < KNN; ++k) {
    if (av[k] > 0.f) {
      int nj = nbr[(size_t)cf * KNN + k];
      out += av[k] * inv *
             (Lb[(size_t)nj * CDIM + c] + delta[(size_t)(rbase + k) * CDIM + c]);
    }
  }
  float mean = blockSum384(out, s) * (1.f / CDIM);
  float d    = out - mean;
  float var  = blockSum384(d * d, s) * (1.f / CDIM);
  Xout[(size_t)cf * CDIM + c] = d * rsqrtf(var + 1e-5f) * g[c] + beta[c];
}

// ---------------------------------------------------------------------------
// Row LayerNorm. grid = rows, block = CDIM. Safe in-place.
// ---------------------------------------------------------------------------
__global__ void layernorm_rows_kernel(const float* __restrict__ Xin, const float* __restrict__ g,
                                      const float* __restrict__ b, float* __restrict__ Xout) {
  __shared__ float s[CDIM];
  int r = blockIdx.x, c = threadIdx.x;
  float v    = Xin[(size_t)r * CDIM + c];
  float mean = blockSum384(v, s) * (1.f / CDIM);
  float d    = v - mean;
  float var  = blockSum384(d * d, s) * (1.f / CDIM);
  Xout[(size_t)r * CDIM + c] = d * rsqrtf(var + 1e-5f) * g[c] + b[c];
}

// ---------------------------------------------------------------------------
// Pooled single-query MHA. grid = B*HEADS, block = 64. Np = 16 keys.
// ---------------------------------------------------------------------------
__global__ void pool_attn_kernel(const float* __restrict__ q, const float* __restrict__ kmat,
                                 const float* __restrict__ vmat, float* __restrict__ obuf,
                                 int Np) {
  __shared__ float sc[16];
  int b = blockIdx.x >> 3;
  int h = blockIdx.x & 7;
  int t = threadIdx.x;
  const int dh = CDIM / 8;                               // 48
  if (t < Np) {
    float sum = 0.f;
    for (int d = 0; d < dh; ++d)
      sum += q[h * dh + d] * kmat[(size_t)(b * Np + t) * CDIM + h * dh + d];
    sc[t] = sum * 0.14433756729740643f;                  // 48^-0.5
  }
  __syncthreads();
  float mx = -3e38f;
  for (int j = 0; j < Np; ++j) mx = fmaxf(mx, sc[j]);
  float se = 0.f;
  for (int j = 0; j < Np; ++j) se += __expf(sc[j] - mx);
  if (t < dh) {
    float o = 0.f;
    for (int j = 0; j < Np; ++j)
      o += __expf(sc[j] - mx) / se * vmat[(size_t)(b * Np + j) * CDIM + h * dh + t];
    obuf[(size_t)b * CDIM + h * dh + t] = o;
  }
}

// ---------------------------------------------------------------------------
extern "C" void kernel_launch(void* const* d_in, const int* in_sizes, int n_in,
                              void* d_out, int out_size, void* d_ws, size_t ws_size,
                              hipStream_t stream) {
  (void)in_sizes; (void)n_in; (void)out_size; (void)ws_size;
  const float* pts        = (const float*)d_in[0];
  const float* pos_w1     = (const float*)d_in[1];
  const float* pos_b1     = (const float*)d_in[2];
  const float* pos_w2     = (const float*)d_in[3];
  const float* pos_b2     = (const float*)d_in[4];
  const float* attn_w     = (const float*)d_in[5];
  const float* attn_b     = (const float*)d_in[6];
  const float* lin_w      = (const float*)d_in[7];
  const float* lin_b      = (const float*)d_in[8];
  const float* src_w      = (const float*)d_in[9];
  const float* src_b      = (const float*)d_in[10];
  const float* dst_w      = (const float*)d_in[11];
  const float* dst_b      = (const float*)d_in[12];
  const float* ln_g       = (const float*)d_in[13];
  const float* ln_bt      = (const float*)d_in[14];
  const float* proj_w     = (const float*)d_in[15];
  const float* proj_b     = (const float*)d_in[16];
  const float* proj_ln_g  = (const float*)d_in[17];
  const float* proj_ln_b  = (const float*)d_in[18];
  const float* pool_query = (const float*)d_in[19];
  const float* wq         = (const float*)d_in[20];
  const float* wk         = (const float*)d_in[21];
  const float* wv         = (const float*)d_in[22];
  const float* wo         = (const float*)d_in[23];
  const float* bq         = (const float*)d_in[24];
  const float* bk         = (const float*)d_in[25];
  const float* bv         = (const float*)d_in[26];
  const float* bo         = (const float*)d_in[27];
  const float* pn_g       = (const float*)d_in[28];
  const float* pn_b       = (const float*)d_in[29];

  const int N0 = 4096;
  size_t off = 0;
  auto carve = [&](size_t nbytes) -> char* {
    char* r = (char*)d_ws + off;
    off += (nbytes + 255) & ~(size_t)255;
    return r;
  };
  float* posA = (float*)carve((size_t)BB * N0 * 3 * 4);
  float* posB = (float*)carve((size_t)BB * (N0 / 4) * 3 * 4);
  float* xA   = (float*)carve((size_t)BB * N0 * CDIM * 4);
  float* xB   = (float*)carve((size_t)BB * (N0 / 4) * CDIM * 4);
  int*   idx  = (int*)carve((size_t)BB * (N0 / 4) * 4);
  int*   nbr  = (int*)carve((size_t)BB * (N0 / 4) * KNN * 4);
  int*   vmsk = (int*)carve((size_t)BB * (N0 / 4) * KNN * 4);
  float* Sbuf = (float*)carve((size_t)BB * N0 * CDIM * 4);
  float* Lbuf = (float*)carve((size_t)BB * N0 * CDIM * 4);
  float* dtrm = (float*)carve((size_t)BB * (N0 / 4) * CDIM * 4);
  const int CHUNK = 256;                                  // centers per chunk
  float* bufH = (float*)carve((size_t)CHUNK * KNN * CDIM * 4);
  float* bufD = (float*)carve((size_t)CHUNK * KNN * CDIM * 4);
  float* bufA = (float*)carve((size_t)CHUNK * KNN * CDIM * 4);
  float* qrow = (float*)carve((size_t)CDIM * 4);
  float* kmat = (float*)carve((size_t)BB * 16 * CDIM * 4);
  float* vmat = (float*)carve((size_t)BB * 16 * CDIM * 4);
  float* obuf = (float*)carve((size_t)BB * CDIM * 4);
  float* ptmp = (float*)carve((size_t)BB * 16 * CDIM * 4);
  float* otmp = (float*)carve((size_t)BB * CDIM * 4);

  auto gemm = [&](const float* X, const float* W, const float* bias, float* Y,
                  int R, int act) {
    dim3 g((unsigned)((R + 31) / 32), CDIM / 64);
    wmma_gemm_kernel<<<g, 128, 0, stream>>>(X, W, bias, Y, R, act);
  };

  split_pts_kernel<<<BB * N0, CDIM, 0, stream>>>(pts, posA, xA);

  float* posIn = posA; float* posOut = posB;
  float* xIn = xA;     float* xOut = xB;
  int Ncur = N0;
  const float radii2[4] = {1.f, 4.f, 16.f, 64.f};

  for (int lvl = 0; lvl < 4; ++lvl) {
    int M  = Ncur / 4;
    int NC = BB * M;
    const float* w1 = pos_w1 + (size_t)lvl * 3 * CDIM;
    const float* b1 = pos_b1 + (size_t)lvl * CDIM;
    const float* w2 = pos_w2 + (size_t)lvl * CDIM * CDIM;
    const float* b2 = pos_b2 + (size_t)lvl * CDIM;
    const float* aw = attn_w + (size_t)lvl * CDIM * CDIM;
    const float* ab = attn_b + (size_t)lvl * CDIM;
    const float* lw = lin_w  + (size_t)lvl * CDIM * CDIM;
    const float* lb = lin_b  + (size_t)lvl * CDIM;
    const float* sw = src_w  + (size_t)lvl * CDIM * CDIM;
    const float* sb = src_b  + (size_t)lvl * CDIM;
    const float* dw = dst_w  + (size_t)lvl * CDIM * CDIM;
    const float* db = dst_b  + (size_t)lvl * CDIM;
    const float* lg = ln_g   + (size_t)lvl * CDIM;
    const float* lbeta = ln_bt + (size_t)lvl * CDIM;

    fps_kernel<<<BB, 256, 0, stream>>>(posIn, idx, Ncur, M);
    gather_kernel<<<NC, CDIM, 0, stream>>>(xIn, posIn, idx, xOut, posOut, M, Ncur);
    knn_kernel<<<(NC + 63) / 64, 64, 0, stream>>>(posIn, posOut, nbr, vmsk, NC, Ncur, M,
                                                  radii2[lvl]);
    gemm(xIn, sw, sb, Sbuf, BB * Ncur, 0);     // S = x @ sw + sb
    gemm(xIn, lw, lb, Lbuf, BB * Ncur, 0);     // L = x @ lw + lb
    gemm(xOut, dw, db, dtrm, NC, 0);           // D = x_d @ dw + db (xOut holds x_d)

    for (int c0 = 0; c0 < NC; c0 += CHUNK) {
      int nc   = (NC - c0 < CHUNK) ? (NC - c0) : CHUNK;
      int rows = nc * KNN;
      posmlp1_kernel<<<rows, CDIM, 0, stream>>>(posIn, posOut, nbr, w1, b1, bufH, c0);
      gemm(bufH, w2, b2, bufD, rows, 1);       // delta = relu(h1 @ w2 + b2)
      combine_kernel<<<rows, CDIM, 0, stream>>>(dtrm, Sbuf, nbr, bufD, bufH, c0);
      gemm(bufH, aw, ab, bufA, rows, 1);       // alpha = relu(a_in @ aw + ab)
      attn_agg_ln_kernel<<<nc, CDIM, 0, stream>>>(bufA, bufD, Lbuf, nbr, vmsk, lg, lbeta,
                                                  xOut, c0);
    }
    { float* t = posIn; posIn = posOut; posOut = t; }
    { float* t = xIn; xIn = xOut; xOut = t; }
    Ncur = M;
  }

  // xIn: (B, 16, C). Projection + LN, pooled attention, output head.
  gemm(xIn, proj_w, proj_b, ptmp, BB * 16, 0);
  layernorm_rows_kernel<<<BB * 16, CDIM, 0, stream>>>(ptmp, proj_ln_g, proj_ln_b, ptmp);
  gemm(pool_query, wq, bq, qrow, 1, 0);        // q = pq @ wq + bq
  gemm(ptmp, wk, bk, kmat, BB * 16, 0);
  gemm(ptmp, wv, bv, vmat, BB * 16, 0);
  pool_attn_kernel<<<BB * 8, 64, 0, stream>>>(qrow, kmat, vmat, obuf, 16);
  gemm(obuf, wo, bo, otmp, BB, 0);
  layernorm_rows_kernel<<<BB, CDIM, 0, stream>>>(otmp, pn_g, pn_b, (float*)d_out);
}
